// EncoderLayer_9216999817377
// MI455X (gfx1250) — matrix operand order repaired and checked by reading, not compiled, not run on hardware
//
#include <hip/hip_runtime.h>

// ---------------------------------------------------------------------------
// MI455X (gfx1250) transformer encoder layer.
// B=4, S=2048, D=768, H=12, Dh=64, Dff=3072.
// - All matmuls via v_wmma_f32_16x16x32_f16 (wave32).
// - Tensor Data Mover double-buffers A/B (GEMM, 128x128 block tile) and K/V
//   (attention) tiles into LDS; fragments via ds_load_b128; s_wait_tensorcnt.
//   One scalar descriptor per tensor; lds/global address patched per issue
//   (no descriptor arrays -> no scratch spills).
// - Flash attention: softmax reductions in A-layout (row = 2 lanes -> in-lane
//   VALU reduction + one xor-16 shuffle), bit-packed mask.
// ---------------------------------------------------------------------------

typedef __attribute__((ext_vector_type(16))) _Float16     v16h;
typedef __attribute__((ext_vector_type(8)))  _Float16     h8;
typedef __attribute__((ext_vector_type(8)))  float        v8f;
typedef __attribute__((ext_vector_type(4)))  unsigned int v4u;
typedef __attribute__((ext_vector_type(8)))  int          v8i;
typedef __attribute__((ext_vector_type(4)))  int          v4i;

#define B_SZ   4
#define S_SZ   2048
#define D_SZ   768
#define H_SZ   12
#define DH_SZ  64
#define DF_SZ  3072
#define NT_SZ  (B_SZ * S_SZ)          // 8192 tokens
#define SW_SZ  (S_SZ / 32)            // mask words per row
#define MASKED_SCORE (-30000.0f)      // f16-representable "-inf"

__device__ __forceinline__ v8f vzero8() {
  v8f z = {0.f, 0.f, 0.f, 0.f, 0.f, 0.f, 0.f, 0.f};
  return z;
}

__device__ __forceinline__ v16h cat16(h8 lo, h8 hi) {
  return __builtin_shufflevector(lo, hi, 0, 1, 2, 3, 4, 5, 6, 7,
                                 8, 9, 10, 11, 12, 13, 14, 15);
}

// A-fragment (16x32 f16): lane row at p, K contiguous; lane half covers k 16*(lane>=16)..+15. k0off = 8*(lane>=16).
__device__ __forceinline__ v16h ld_a32(const _Float16* p, int k0off) {
  return cat16(*(const h8*)(p + 2 * k0off), *(const h8*)(p + 2 * k0off + 8));
}

// B-fragment (32x16 f16): lane column at p, K contiguous; caller adds hf*16.
__device__ __forceinline__ v16h ld_b16(const _Float16* p) {
  return cat16(*(const h8*)p, *(const h8*)(p + 8));
}

// ---------------------------------------------------------------------------
// Tensor Data Mover descriptor (CDNA5 ISA 8.3/8.4).  One scalar descriptor
// per tensor; lds_addr and global address patched at issue time so the
// double-buffer parity is a cheap s_cselect, never an indexed struct array.
// ---------------------------------------------------------------------------
struct TdmDesc { v4u g0; v8i g1; };

__device__ __forceinline__ TdmDesc tdm_make_2d(unsigned tile_d0, unsigned tile_d1,
                                               unsigned long long tensor_d0,
                                               unsigned long long tensor_d1,
                                               unsigned long long row_stride) {
  TdmDesc d;
  d.g0.x = 1u;                                            // count=1, user mode
  d.g0.y = 0u;                                            // lds_addr (patched)
  d.g0.z = 0u;                                            // gaddr lo (patched)
  d.g0.w = (2u << 30);                                    // type=2 (+gaddr hi)
  const unsigned d0 = (unsigned)tensor_d0, d1 = (unsigned)tensor_d1;
  d.g1[0] = (int)(1u << 16);                              // data_size=1 (2B)
  d.g1[1] = (int)((d0 & 0xFFFFu) << 16);                  // tensor_dim0 lo
  d.g1[2] = (int)((d0 >> 16) | ((d1 & 0xFFFFu) << 16));   // dim0 hi | dim1 lo
  d.g1[3] = (int)((d1 >> 16) | (tile_d0 << 16));          // dim1 hi | tile0
  d.g1[4] = (int)(tile_d1 & 0xFFFFu);                     // tile1, tile2=0
  d.g1[5] = (int)(row_stride & 0xFFFFFFFFull);            // dim0_stride lo
  d.g1[6] = (int)((row_stride >> 32) & 0xFFFFull);        // dim0_stride hi
  d.g1[7] = 0;
  return d;
}

__device__ __forceinline__ void tdm_issue(const TdmDesc& d, unsigned lds_off,
                                          const void* gptr) {
  const unsigned long long ga = (unsigned long long)gptr;
  v4u g0 = d.g0;
  g0.y = lds_off;
  g0.z = (unsigned)ga;
  g0.w = d.g0.w | (unsigned)((ga >> 32) & 0x01FFFFFFull);
  v4i gz4 = {0, 0, 0, 0};
  v8i gz8 = {0, 0, 0, 0, 0, 0, 0, 0};
  __builtin_amdgcn_tensor_load_to_lds(g0, d.g1, gz4, gz4, gz8, 0);
}

__device__ __forceinline__ unsigned lds_off_of(const void* p) {
  return (unsigned)(unsigned long long)p;   // LDS offset is the low 32 bits of the address
}

// ---------------------------------------------------------------------------
__global__ __launch_bounds__(256) void cast_f16_kernel(
    const float* __restrict__ in, _Float16* __restrict__ out, size_t n) {
  for (size_t i = (size_t)blockIdx.x * 256 + threadIdx.x; i < n;
       i += (size_t)gridDim.x * 256)
    out[i] = (_Float16)in[i];
}

// W[k][n] (f32 KxN) -> Wt[n][k] (f16)
__global__ __launch_bounds__(256) void transpose_cast_kernel(
    const float* __restrict__ W, _Float16* __restrict__ Wt, int K, int N) {
  size_t total = (size_t)K * N;
  for (size_t i = (size_t)blockIdx.x * 256 + threadIdx.x; i < total;
       i += (size_t)gridDim.x * 256) {
    size_t k = i / N, n = i % N;
    Wt[n * (size_t)K + k] = (_Float16)W[i];
  }
}

// bool mask [B,S,S] (u8) -> bit-packed [B,S,S/32] (u32, bit j = key j)
__global__ __launch_bounds__(256) void pack_mask_kernel(
    const unsigned char* __restrict__ m, unsigned* __restrict__ out) {
  const size_t nw = (size_t)B_SZ * S_SZ * SW_SZ;
  for (size_t i = (size_t)blockIdx.x * 256 + threadIdx.x; i < nw;
       i += (size_t)gridDim.x * 256) {
    const uint4* p = (const uint4*)(m + i * 32);
    const uint4 a = p[0], b4 = p[1];
    const unsigned wv[8] = {a.x, a.y, a.z, a.w, b4.x, b4.y, b4.z, b4.w};
    unsigned bits = 0;
#pragma unroll
    for (int j = 0; j < 8; j++)
#pragma unroll
      for (int c = 0; c < 4; c++)
        if ((wv[j] >> (8 * c)) & 0xFFu) bits |= 1u << (j * 4 + c);
    out[i] = bits;
  }
}

// ---------------------------------------------------------------------------
// WMMA GEMM:  C[M][N] = A[M][K](f16 rm) * Bt[N][K]^T + bias
// 8 waves; wave tile 32x64 (2x4 WMMA); block tile 128x128; TDM double-buffer.
// MODE 0: f16 rm  1: f32 rm  2: f16 [B,H,S,Dh]  3: f16 [B,H,Dh,S]
// ---------------------------------------------------------------------------
template <int MODE>
__global__ __launch_bounds__(256) void gemm_wmma_kernel(
    const _Float16* __restrict__ A, const _Float16* __restrict__ Bt,
    const float* __restrict__ bias, void* __restrict__ outv,
    int M, int N, int K) {
  __shared__ __align__(16) _Float16 As[2][128][32];
  __shared__ __align__(16) _Float16 Bs[2][128][32];

  const int lane = threadIdx.x & 31, w = threadIdx.x >> 5;
  const int lm = lane & 15, hf = lane >> 4, k0off = hf * 8;
  const int mBlk = blockIdx.y * 128, nBlk = blockIdx.x * 128;
  const int mW = (w & 3) * 32, nW = (w >> 2) * 64;

  v8f c[2][4];
#pragma unroll
  for (int mi = 0; mi < 2; mi++)
#pragma unroll
    for (int ni = 0; ni < 4; ni++) c[mi][ni] = vzero8();

  // Uniform scalar descriptors (computed by every wave, issued by wave 0).
  const TdmDesc dA = tdm_make_2d(32, 128, K, M, K);
  const TdmDesc dB = tdm_make_2d(32, 128, K, N, K);
  const unsigned offA0 = lds_off_of(&As[0][0][0]);
  const unsigned offA1 = lds_off_of(&As[1][0][0]);
  const unsigned offB0 = lds_off_of(&Bs[0][0][0]);
  const unsigned offB1 = lds_off_of(&Bs[1][0][0]);

  if (w == 0) {
    tdm_issue(dA, offA0, A + (size_t)mBlk * K);
    tdm_issue(dB, offB0, Bt + (size_t)nBlk * K);
  }

  const int iters = K / 32;
  for (int it = 0; it < iters; ++it) {
    const int k = it * 32, par = it & 1;
    if (w == 0) {
      if (it + 1 < iters) {
        tdm_issue(dA, par ? offA0 : offA1, A + (size_t)mBlk * K + k + 32);
        tdm_issue(dB, par ? offB0 : offB1, Bt + (size_t)nBlk * K + k + 32);
        __builtin_amdgcn_s_wait_tensorcnt(2);   // current buffers complete
      } else {
        __builtin_amdgcn_s_wait_tensorcnt(0);
      }
    }
    __syncthreads();                            // buffers[par] visible

    v16h a[2], b[4];
#pragma unroll
    for (int mi = 0; mi < 2; mi++)
      a[mi] = ld_a32(&As[par][mW + mi * 16 + lm][0], k0off);
#pragma unroll
    for (int ni = 0; ni < 4; ni++)
      b[ni] = ld_b16(&Bs[par][nW + ni * 16 + lm][hf * 16]);
#pragma unroll
    for (int mi = 0; mi < 2; mi++)
#pragma unroll
      for (int ni = 0; ni < 4; ni++)
        c[mi][ni] = __builtin_amdgcn_wmma_f32_16x16x32_f16(
            false, a[mi], false, b[ni], (short)0, c[mi][ni], false, false);

    __syncthreads();                            // done reading buffers[par]
  }

#pragma unroll
  for (int mi = 0; mi < 2; mi++)
#pragma unroll
    for (int ni = 0; ni < 4; ni++) {
      const int n = nBlk + nW + ni * 16 + lm;
      const float bv = bias[n];
#pragma unroll
      for (int i = 0; i < 8; i++) {
        const int m = mBlk + mW + mi * 16 + i + 8 * hf;
        const float val = c[mi][ni][i] + bv;
        if constexpr (MODE == 0) {
          ((_Float16*)outv)[(size_t)m * N + n] = (_Float16)val;
        } else if constexpr (MODE == 1) {
          ((float*)outv)[(size_t)m * N + n] = val;
        } else if constexpr (MODE == 2) {
          const int bb = m >> 11, ss = m & 2047, hh = n >> 6, dd = n & 63;
          ((_Float16*)outv)[(((size_t)(bb * H_SZ + hh) * S_SZ) + ss) * DH_SZ + dd] =
              (_Float16)val;
        } else {
          const int bb = m >> 11, ss = m & 2047, hh = n >> 6, dd = n & 63;
          ((_Float16*)outv)[(((size_t)(bb * H_SZ + hh) * DH_SZ) + dd) * S_SZ + ss] =
              (_Float16)val;
        }
      }
    }
}

// ---------------------------------------------------------------------------
// Flash attention.  grid = (S/128, B*H); 8 waves; wave = 16 query rows.
// q,k: [B,H,S,64] f16; vt: [B,H,64,S] f16; maskp: bit-packed [B,S,S/32].
// Scores: C-layout -> LDS -> A-layout (row = lanes {r, r+16}) so softmax
// max/sum are in-lane VALU reductions + ONE xor-16 shuffle each.  exp() in
// registers; result is directly the P fragment for the PV WMMA.  Row stats
// broadcast back to C-layout accumulators through a tiny LDS array.
// ---------------------------------------------------------------------------
__global__ __launch_bounds__(256) void attn_flash_kernel(
    const _Float16* __restrict__ q, const _Float16* __restrict__ k,
    const _Float16* __restrict__ vt, const unsigned* __restrict__ maskp,
    float* __restrict__ out) {
  __shared__ __align__(16) _Float16 Ks[2][32][64];   // [key][depth]
  __shared__ __align__(16) _Float16 Vs[2][64][32];   // [depth][key]
  __shared__ __align__(16) _Float16 Pl[8][16][32];   // per-wave score staging
  __shared__ float Abuf[8][16];                      // per-wave row-stat bcast

  const int lane = threadIdx.x & 31, w = threadIdx.x >> 5;
  const int lm = lane & 15, hf = lane >> 4, k0off = hf * 8;
  const int bh = blockIdx.y;
  const int bt = bh / H_SZ, hh = bh % H_SZ;
  const int qBase = blockIdx.x * 128 + w * 16;

  const _Float16* qp = q + (size_t)bh * S_SZ * DH_SZ;
  const _Float16* kp = k + (size_t)bh * S_SZ * DH_SZ;
  const _Float16* vp = vt + (size_t)bh * DH_SZ * S_SZ;
  const unsigned* mpk = maskp + (size_t)bt * S_SZ * SW_SZ;

  // Q fragments (16 q x 64 depth = two 16x32 A tiles), loaded once.
  const _Float16* qrow = qp + (size_t)(qBase + lm) * DH_SZ;
  const v16h aq0 = ld_a32(qrow, k0off);
  const v16h aq1 = ld_a32(qrow + 32, k0off);

  v8f o[4];
#pragma unroll
  for (int j = 0; j < 4; j++) o[j] = vzero8();
  float mstat = -3.0e30f, lstat = 0.f;   // stats for query row (qBase+lm)

  const TdmDesc dK = tdm_make_2d(64, 32, DH_SZ, S_SZ, DH_SZ);
  const TdmDesc dV = tdm_make_2d(32, 64, S_SZ, DH_SZ, S_SZ);
  const unsigned offK0 = lds_off_of(&Ks[0][0][0]);
  const unsigned offK1 = lds_off_of(&Ks[1][0][0]);
  const unsigned offV0 = lds_off_of(&Vs[0][0][0]);
  const unsigned offV1 = lds_off_of(&Vs[1][0][0]);

  if (w == 0) {
    tdm_issue(dK, offK0, kp);
    tdm_issue(dV, offV0, vp);
  }

  for (int kc = 0; kc < S_SZ; kc += 32) {
    const int par = (kc >> 5) & 1;
    if (w == 0) {
      if (kc + 32 < S_SZ) {
        tdm_issue(dK, par ? offK0 : offK1, kp + (size_t)(kc + 32) * DH_SZ);
        tdm_issue(dV, par ? offV0 : offV1, vp + kc + 32);
        __builtin_amdgcn_s_wait_tensorcnt(2);
      } else {
        __builtin_amdgcn_s_wait_tensorcnt(0);
      }
    }
    __syncthreads();

    // ---- scores: S[16 q][32 keys], 4 WMMAs from LDS K tiles ----
    v8f s[2];
#pragma unroll
    for (int t = 0; t < 2; t++) {
      const _Float16* kr = &Ks[par][16 * t + lm][hf * 16];
      const v16h bk0 = ld_b16(kr);        // depth 0..31
      const v16h bk1 = ld_b16(kr + 32);   // depth 32..63
      v8f acc = vzero8();
      acc = __builtin_amdgcn_wmma_f32_16x16x32_f16(false, aq0, false, bk0,
                                                   (short)0, acc, false, false);
      acc = __builtin_amdgcn_wmma_f32_16x16x32_f16(false, aq1, false, bk1,
                                                   (short)0, acc, false, false);
      s[t] = acc;
    }
    // ---- scale + mask in C layout; stage raw scores to LDS ----
#pragma unroll
    for (int i = 0; i < 8; i++) {
      const int qi = qBase + i + 8 * hf;
      const unsigned mw = mpk[(size_t)qi * SW_SZ + (kc >> 5)];
      float s0v = s[0][i] * 0.125f;       // 1/sqrt(64)
      float s1v = s[1][i] * 0.125f;
      if ((mw >> lm) & 1u) s0v = MASKED_SCORE;
      if ((mw >> (lm + 16)) & 1u) s1v = MASKED_SCORE;
      Pl[w][i + 8 * hf][lm]      = (_Float16)s0v;
      Pl[w][i + 8 * hf][lm + 16] = (_Float16)s1v;
    }
    asm volatile("s_wait_dscnt 0" ::: "memory");
    // ---- A-layout softmax: row (qBase+lm) in lanes {lm, lm+16} ----
    const v16h raw = ld_a32(&Pl[w][lm][0], k0off);
    float mx = -3.0e30f;
#pragma unroll
    for (int h = 0; h < 16; h++) mx = fmaxf(mx, (float)raw[h]);
    mx = fmaxf(mx, __shfl_xor(mx, 16, 32));
    const float mn = fmaxf(mstat, mx);
    const float alpha = __expf(mstat - mn);
    float rs = 0.f;
    v16h pf;
#pragma unroll
    for (int h = 0; h < 16; h++) {
      const float e = __expf((float)raw[h] - mn);
      rs += e;
      pf[h] = (_Float16)e;
    }
    rs += __shfl_xor(rs, 16, 32);
    lstat = lstat * alpha + rs;
    mstat = mn;
    // ---- broadcast alpha back to C-layout rows, rescale O ----
    if (hf == 0) Abuf[w][lm] = alpha;
    asm volatile("s_wait_dscnt 0" ::: "memory");
#pragma unroll
    for (int i = 0; i < 8; i++) {
      const float av = Abuf[w][i + 8 * hf];
#pragma unroll
      for (int j = 0; j < 4; j++) o[j][i] *= av;
    }
    // ---- O(16x64) += P(16x32) x V(32x64); pf is already the A fragment ----
#pragma unroll
    for (int j = 0; j < 4; j++) {
      const v16h bv = ld_b16(&Vs[par][16 * j + lm][hf * 16]);
      o[j] = __builtin_amdgcn_wmma_f32_16x16x32_f16(false, pf, false, bv,
                                                    (short)0, o[j], false, false);
    }
    __syncthreads();
  }

  // ---- normalize (broadcast 1/l) + write ctx as [B,S,H*Dh] f32 ----
  if (hf == 0) Abuf[w][lm] = 1.0f / lstat;
  asm volatile("s_wait_dscnt 0" ::: "memory");
#pragma unroll
  for (int j = 0; j < 4; j++)
#pragma unroll
    for (int i = 0; i < 8; i++) {
      const int qi = qBase + i + 8 * hf;
      out[((size_t)bt * S_SZ + qi) * D_SZ + hh * DH_SZ + 16 * j + lm] =
          o[j][i] * Abuf[w][i + 8 * hf];
    }
}

// ---------------------------------------------------------------------------
// y = LayerNorm(x + r) * g + beta ; f32 out (+ optional f16 copy).
// ---------------------------------------------------------------------------
__global__ __launch_bounds__(256) void add_ln_kernel(
    const float* __restrict__ x, const float* __restrict__ r,
    const float* __restrict__ g, const float* __restrict__ be,
    float* __restrict__ out32, _Float16* __restrict__ out16) {
  __shared__ float sred[8];
  const int row = blockIdx.x, tix = threadIdx.x;
  const int w = tix >> 5, lane = tix & 31;
  const float* xr = x + (size_t)row * D_SZ;
  const float* rr = r + (size_t)row * D_SZ;

  float t[3], s = 0.f;
#pragma unroll
  for (int e = 0; e < 3; e++) {
    t[e] = xr[tix + 256 * e] + rr[tix + 256 * e];
    s += t[e];
  }
#pragma unroll
  for (int o = 1; o < 32; o <<= 1) s += __shfl_xor(s, o, 32);
  if (lane == 0) sred[w] = s;
  __syncthreads();
  float tot = 0.f;
#pragma unroll
  for (int i = 0; i < 8; i++) tot += sred[i];
  const float mu = tot * (1.0f / D_SZ);
  __syncthreads();

  float s2 = 0.f;
#pragma unroll
  for (int e = 0; e < 3; e++) { const float d = t[e] - mu; s2 += d * d; }
#pragma unroll
  for (int o = 1; o < 32; o <<= 1) s2 += __shfl_xor(s2, o, 32);
  if (lane == 0) sred[w] = s2;
  __syncthreads();
  float var = 0.f;
#pragma unroll
  for (int i = 0; i < 8; i++) var += sred[i];
  const float rstd = rsqrtf(var * (1.0f / D_SZ) + 1e-5f);

#pragma unroll
  for (int e = 0; e < 3; e++) {
    const int c = tix + 256 * e;
    const float y = (t[e] - mu) * rstd * g[c] + be[c];
    out32[(size_t)row * D_SZ + c] = y;
    if (out16) out16[(size_t)row * D_SZ + c] = (_Float16)y;
  }
}

// ---------------------------------------------------------------------------
extern "C" void kernel_launch(void* const* d_in, const int* in_sizes, int n_in,
                              void* d_out, int out_size, void* d_ws,
                              size_t ws_size, hipStream_t stream) {
  (void)in_sizes; (void)n_in; (void)out_size; (void)ws_size;

  const float*         x_in = (const float*)d_in[0];
  const unsigned char* mask = (const unsigned char*)d_in[1];   // jax bool
  const float* Wq = (const float*)d_in[2];  const float* bq = (const float*)d_in[3];
  const float* Wk = (const float*)d_in[4];  const float* bk = (const float*)d_in[5];
  const float* Wv = (const float*)d_in[6];  const float* bv = (const float*)d_in[7];
  const float* g1 = (const float*)d_in[8];  const float* be1 = (const float*)d_in[9];
  const float* W1 = (const float*)d_in[10]; const float* b1 = (const float*)d_in[11];
  const float* W2 = (const float*)d_in[12]; const float* b2 = (const float*)d_in[13];
  const float* g2 = (const float*)d_in[14]; const float* be2 = (const float*)d_in[15];

  char* ws = (char*)d_ws;
  size_t off = 0;
  auto alloc = [&](size_t bytes) {
    size_t r = off;
    off += (bytes + 255) & ~(size_t)255;
    return r;
  };
  _Float16* X16   = (_Float16*)(ws + alloc((size_t)NT_SZ * D_SZ * 2));
  _Float16* WqT   = (_Float16*)(ws + alloc((size_t)D_SZ * D_SZ * 2));
  _Float16* WkT   = (_Float16*)(ws + alloc((size_t)D_SZ * D_SZ * 2));
  _Float16* WvT   = (_Float16*)(ws + alloc((size_t)D_SZ * D_SZ * 2));
  _Float16* W1T   = (_Float16*)(ws + alloc((size_t)D_SZ * DF_SZ * 2));
  _Float16* W2T   = (_Float16*)(ws + alloc((size_t)DF_SZ * D_SZ * 2));
  _Float16* qh    = (_Float16*)(ws + alloc((size_t)NT_SZ * D_SZ * 2));
  _Float16* khb   = (_Float16*)(ws + alloc((size_t)NT_SZ * D_SZ * 2));
  _Float16* vth   = (_Float16*)(ws + alloc((size_t)NT_SZ * D_SZ * 2));
  unsigned* maskp = (unsigned*)(ws + alloc((size_t)B_SZ * S_SZ * SW_SZ * 4));
  float*    attn  = (float*)(ws + alloc((size_t)NT_SZ * D_SZ * 4));
  float*    fin   = (float*)(ws + alloc((size_t)NT_SZ * D_SZ * 4));
  _Float16* fin16 = (_Float16*)(ws + alloc((size_t)NT_SZ * D_SZ * 2));
  _Float16* hid16 = (_Float16*)(ws + alloc((size_t)NT_SZ * DF_SZ * 2));
  float*    fout  = (float*)(ws + alloc((size_t)NT_SZ * D_SZ * 4));

  const dim3 blk(256);

  // 1) casts / transposes / mask packing
  cast_f16_kernel<<<2048, blk, 0, stream>>>(x_in, X16, (size_t)NT_SZ * D_SZ);
  transpose_cast_kernel<<<1024, blk, 0, stream>>>(Wq, WqT, D_SZ, D_SZ);
  transpose_cast_kernel<<<1024, blk, 0, stream>>>(Wk, WkT, D_SZ, D_SZ);
  transpose_cast_kernel<<<1024, blk, 0, stream>>>(Wv, WvT, D_SZ, D_SZ);
  transpose_cast_kernel<<<2048, blk, 0, stream>>>(W1, W1T, D_SZ, DF_SZ);
  transpose_cast_kernel<<<2048, blk, 0, stream>>>(W2, W2T, DF_SZ, D_SZ);
  pack_mask_kernel<<<2048, blk, 0, stream>>>(mask, maskp);

  // 2) QKV projections: Q/K -> [B,H,S,Dh], V -> [B,H,Dh,S]
  gemm_wmma_kernel<2><<<dim3(D_SZ / 128, NT_SZ / 128), blk, 0, stream>>>(
      X16, WqT, bq, qh, NT_SZ, D_SZ, D_SZ);
  gemm_wmma_kernel<2><<<dim3(D_SZ / 128, NT_SZ / 128), blk, 0, stream>>>(
      X16, WkT, bk, khb, NT_SZ, D_SZ, D_SZ);
  gemm_wmma_kernel<3><<<dim3(D_SZ / 128, NT_SZ / 128), blk, 0, stream>>>(
      X16, WvT, bv, vth, NT_SZ, D_SZ, D_SZ);

  // 3) Flash attention -> attn [B,S,D] f32
  attn_flash_kernel<<<dim3(S_SZ / 128, B_SZ * H_SZ), blk, 0, stream>>>(
      qh, khb, vth, maskp, attn);

  // 4) Add & Norm 1
  add_ln_kernel<<<NT_SZ, blk, 0, stream>>>(attn, x_in, g1, be1, fin, fin16);

  // 5) FFN (no activation per reference)
  gemm_wmma_kernel<0><<<dim3(DF_SZ / 128, NT_SZ / 128), blk, 0, stream>>>(
      fin16, W1T, b1, hid16, NT_SZ, DF_SZ, D_SZ);
  gemm_wmma_kernel<1><<<dim3(D_SZ / 128, NT_SZ / 128), blk, 0, stream>>>(
      hid16, W2T, b2, fout, NT_SZ, D_SZ, DF_SZ);

  // 6) Add & Norm 2 -> output (f32)
  add_ln_kernel<<<NT_SZ, blk, 0, stream>>>(fout, fin, g2, be2, (float*)d_out,
                                           nullptr);
}
